// BioDecoder_18769007084010
// MI455X (gfx1250) — matrix operand order repaired and checked by reading, not compiled, not run on hardware
//
#include <hip/hip_runtime.h>
#include <math.h>

namespace {

constexpr int NB = 64;      // batch
constexpr int NE = 512;     // embed
constexpr int NH = 1024;    // hidden
constexpr int NV = 32000;   // vocab
constexpr int NT = 64;      // max_len
constexpr int NG = 4 * NH;  // 4096 gates
constexpr float INV_TEMP = 1.0f / 0.9f;

typedef __bf16 bf16;
typedef __attribute__((ext_vector_type(16))) __bf16 v16bf;
typedef __attribute__((ext_vector_type(8)))  float  v8f;

__device__ __forceinline__ float sigm(float x) { return 1.0f / (1.0f + __expf(-x)); }

__device__ __forceinline__ unsigned hash_u32(unsigned x) {
  x ^= x >> 16; x *= 0x7feb352dU; x ^= x >> 15; x *= 0x846ca68bU; x ^= x >> 16;
  return x;
}

// ---------------- WMMA fragment loaders (wave32, 16x16x32 bf16) ----------------
// A tile: M=16 x K=32 from row-major bf16 buffer. Layout used here: lanes 0-15 hold rows,
// half-wave selects K 0-15 vs 16-31, K contiguous per lane.
__device__ __forceinline__ v16bf load_a(const bf16* __restrict__ A, int lda,
                                        int row0, int k0, int lane) {
  const bf16* p = A + (size_t)(row0 + (lane & 15)) * lda + k0 + ((lane >> 4) << 4);
  v16bf a;
#pragma unroll
  for (int i = 0; i < 16; ++i) a[i] = p[i];
  return a;
}

// B tile: K=32 x N=16 with B[k][n] = W[n0+n][k0+k], W row-major (fp32 converted
// in-register, or native bf16). Layout used here: lane%16 = column, lane/16 selects
// K 0-15 vs 16-31, K contiguous per lane.
template <typename WT>
__device__ __forceinline__ v16bf load_b(const WT* __restrict__ W, int ldw,
                                        int n0, int k0, int lane) {
  const WT* p = W + (size_t)(n0 + (lane & 15)) * ldw + k0 + ((lane >> 4) << 4);
  v16bf b;
#pragma unroll
  for (int j = 0; j < 16; ++j) b[j] = (bf16)p[j];
  return b;
}

// C = A1@W1^T (+ A2@W2^T) + bias1 (+ bias2), then epilogue:
//   act==0: none; act==1: tanh; act==2: logits fuse (count penalty + 1/temp + bonus)
// Block = 256 threads = 8 waves: 4 M-tiles x 2 N-tiles -> 64 x 32 outputs per block.
template <typename WT>
__global__ __launch_bounds__(256) void k_gemm(
    const bf16* __restrict__ A1, int K1, const WT* __restrict__ W1,
    const bf16* __restrict__ A2, int K2, const WT* __restrict__ W2,
    const float* __restrict__ bias1, const float* __restrict__ bias2,
    const float* __restrict__ counts,
    float* __restrict__ C, bf16* __restrict__ Cbf, int N, int act) {
  const int lane = threadIdx.x & 31;
  const int wave = threadIdx.x >> 5;
  const int m0 = (wave & 3) << 4;
  const int n0 = blockIdx.x * 32 + ((wave >> 2) << 4);

  v8f acc = {};
  for (int k = 0; k < K1; k += 32) {
    if (k + 32 < K1)  // prefetch next K-tile of this weight row (global_prefetch_b8)
      __builtin_prefetch(W1 + (size_t)(n0 + (lane & 15)) * K1 + k + 32, 0, 1);
    v16bf a = load_a(A1, K1, m0, k, lane);
    v16bf b = load_b(W1, K1, n0, k, lane);
    acc = __builtin_amdgcn_wmma_f32_16x16x32_bf16(false, a, false, b,
                                                  (short)0, acc, false, false);
  }
  if (A2) {
    for (int k = 0; k < K2; k += 32) {
      v16bf a = load_a(A2, K2, m0, k, lane);
      v16bf b = load_b(W2, K2, n0, k, lane);
      acc = __builtin_amdgcn_wmma_f32_16x16x32_bf16(false, a, false, b,
                                                    (short)0, acc, false, false);
    }
  }

  const int col = n0 + (lane & 15);
  float bs = 0.0f;
  if (bias1) bs += bias1[col];
  if (bias2) bs += bias2[col];
  float cnt = 0.0f, bonus = 0.0f, cpen = 0.0f;
  if (act == 2) {
    cnt = counts[col];
    cpen = (cnt > 1.0f) ? 2.0f * cnt : 0.0f;
    bonus = 0.3f / (cnt + 1.0f);
  }
#pragma unroll
  for (int r = 0; r < 8; ++r) {
    const int row = m0 + r + ((lane >> 4) << 3);  // ISA C layout: vgpr r -> M=r(+8)
    float v = acc[r] + bs;
    if (act == 1) v = tanhf(v);
    if (act == 2) v = (v - cpen) * INV_TEMP + bonus;
    C[(size_t)row * N + col] = v;
    if (Cbf) Cbf[(size_t)row * N + col] = (bf16)v;
  }
}

// ---------------- one-time fp32 -> bf16 weight conversion ----------------
__global__ void k_cvt(const float* __restrict__ src, bf16* __restrict__ dst, int n) {
  int idx = blockIdx.x * blockDim.x + threadIdx.x;
  if (idx < n) dst[idx] = (bf16)src[idx];
}

// ---------------- init carry state ----------------
__global__ void k_init(const float* __restrict__ thought, float* h1, float* c1,
                       float* h2, float* c2, float* prev_ctx, int* word,
                       float* counts, int* hist) {
  int idx = blockIdx.x * blockDim.x + threadIdx.x;
  if (idx < NB * NH) {
    float v = thought[idx];
    h1[idx] = v; h2[idx] = v; prev_ctx[idx] = v;
    c1[idx] = 0.0f; c2[idx] = 0.0f;
  }
  if (idx < NV) counts[idx] = 0.0f;
  if (idx < NT * NB) hist[idx] = 0;
  if (idx < NB) word[idx] = 1;
}

// ---------------- per-step input prep: x0 = [emb[word] | prev_ctx], bf16 states ----
__global__ void k_prep(const float* __restrict__ emb, const int* __restrict__ word,
                       const float* __restrict__ prev_ctx,
                       const float* __restrict__ h1, const float* __restrict__ h2,
                       bf16* x0, bf16* h1b, bf16* h2b) {
  int idx = blockIdx.x * blockDim.x + threadIdx.x;
  const int NX = NB * (NE + NH);  // 98304
  if (idx < NX) {
    int b = idx / (NE + NH), j = idx % (NE + NH);
    float v = (j < NE) ? emb[(size_t)word[b] * NE + j]
                       : prev_ctx[b * NH + (j - NE)];
    x0[idx] = (bf16)v;
  } else {
    int r = idx - NX;
    if (r < NB * NH) h1b[r] = (bf16)h1[r];
    else {
      r -= NB * NH;
      if (r < NB * NH) h2b[r] = (bf16)h2[r];
    }
  }
}

// ---------------- LSTM pointwise (torch gate order i,f,g,o) ----------------
__global__ void k_lstm_elem(const float* __restrict__ gates, float* h, float* c,
                            bf16* hb) {
  int idx = blockIdx.x * blockDim.x + threadIdx.x;  // NB*NH
  int b = idx >> 10, j = idx & (NH - 1);
  const float* g = gates + (size_t)b * NG;
  float i_ = sigm(g[j]);
  float f_ = sigm(g[NH + j]);
  float gg = tanhf(g[2 * NH + j]);
  float o_ = sigm(g[3 * NH + j]);
  float cn = f_ * c[idx] + i_ * gg;
  float hn = o_ * tanhf(cn);
  c[idx] = cn; h[idx] = hn; hb[idx] = (bf16)hn;
}

// ---------------- gate scalar: sigmoid(tanh(h2@gw1^T+gb1)@gw2^T+gb2), out=h2*gate ----
__global__ __launch_bounds__(256) void k_gatedot(
    const bf16* __restrict__ tmpb, const float* __restrict__ gw2,
    const float* __restrict__ gb2, const float* __restrict__ h2,
    float* prev_ctx, bf16* outb) {
  int b = blockIdx.x;
  __shared__ float red[256];
  float s = 0.0f;
  for (int i = threadIdx.x; i < NH; i += 256) s += (float)tmpb[b * NH + i] * gw2[i];
  red[threadIdx.x] = s; __syncthreads();
  for (int st = 128; st > 0; st >>= 1) {
    if (threadIdx.x < st) red[threadIdx.x] += red[threadIdx.x + st];
    __syncthreads();
  }
  float gate = sigm(red[0] + gb2[0]);
  for (int i = threadIdx.x; i < NH; i += 256) {
    float v = h2[b * NH + i] * gate;
    prev_ctx[b * NH + i] = v;
    outb[b * NH + i] = (bf16)v;
  }
}

// -------- recency-decayed repetition penalty, pre-divided by temperature --------
// (l - ph - pc)/tau + bonus == [(l - pc)/tau + bonus] - ph/tau : epilogue did [..].
__global__ void k_hist_pen(float* __restrict__ logits, const int* __restrict__ hist,
                           int t) {
  int b = threadIdx.x;  // one thread owns one batch row; deterministic serial adds
  float* L = logits + (size_t)b * NV;
  for (int j = 0; j < t; ++j) {
    int tok = hist[j * NB + b];
    float w = 2.5f * powf(0.8f, (float)(t - j)) * INV_TEMP;
    L[tok] -= w;
  }
}

// ---------------- top-k=60 -> nucleus(0.88) -> Gumbel categorical ----------------
__global__ __launch_bounds__(256) void k_sample(
    const float* __restrict__ logits, float* counts, int* hist, int* word,
    int* __restrict__ tokens_out, int t) {
  constexpr int TOPK = 60, CAP = 192;
  int b = blockIdx.x;
  const float* L = logits + (size_t)b * NV;

  __shared__ float red[256];
  __shared__ float s_lo, s_hi;
  __shared__ int s_n;
  __shared__ float cand_v[CAP];
  __shared__ int cand_i[CAP];

  // 1) row max
  float m = -1e30f;
  for (int i = threadIdx.x; i < NV; i += 256) m = fmaxf(m, L[i]);
  red[threadIdx.x] = m; __syncthreads();
  for (int st = 128; st > 0; st >>= 1) {
    if (threadIdx.x < st) red[threadIdx.x] = fmaxf(red[threadIdx.x], red[threadIdx.x + st]);
    __syncthreads();
  }
  if (threadIdx.x == 0) { s_hi = red[0]; s_lo = red[0] - 80.0f; }
  __syncthreads();

  // 2) binary search threshold so that count(L >= s_hi) <= TOPK (invariant)
  for (int it = 0; it < 24; ++it) {
    float mid = 0.5f * (s_lo + s_hi);
    int c = 0;
    for (int i = threadIdx.x; i < NV; i += 256) c += (L[i] >= mid) ? 1 : 0;
    red[threadIdx.x] = (float)c; __syncthreads();
    for (int st = 128; st > 0; st >>= 1) {
      if (threadIdx.x < st) red[threadIdx.x] += red[threadIdx.x + st];
      __syncthreads();
    }
    if (threadIdx.x == 0) {
      if ((int)red[0] > TOPK) s_lo = mid; else s_hi = mid;
    }
    __syncthreads();
  }

  // 3) gather survivors
  if (threadIdx.x == 0) s_n = 0;
  __syncthreads();
  float thr = s_hi;
  for (int i = threadIdx.x; i < NV; i += 256) {
    float v = L[i];
    if (v >= thr) {
      int p = atomicAdd(&s_n, 1);
      if (p < CAP) { cand_v[p] = v; cand_i[p] = i; }
    }
  }
  __syncthreads();
  int n = s_n < CAP ? s_n : CAP;

  // 4) odd-even sort descending (tie-break ascending index -> deterministic)
  for (int pass = 0; pass < CAP; ++pass) {
    int i = threadIdx.x * 2 + (pass & 1);
    if (i + 1 < n) {
      float va = cand_v[i], vb = cand_v[i + 1];
      int ia = cand_i[i], ib = cand_i[i + 1];
      if (va < vb || (va == vb && ia > ib)) {
        cand_v[i] = vb; cand_v[i + 1] = va;
        cand_i[i] = ib; cand_i[i + 1] = ia;
      }
    }
    __syncthreads();
  }

  // 5) nucleus cutoff + Gumbel argmax (serial over <=60 items)
  if (threadIdx.x == 0) {
    int kk = n < TOPK ? n : TOPK;
    if (kk < 1) kk = 1;
    float mx = cand_v[0];
    float Z = 0.0f;
    for (int i = 0; i < kk; ++i) { red[i] = __expf(cand_v[i] - mx); Z += red[i]; }
    int keep = kk;
    float cum = 0.0f;
    for (int i = 0; i < kk; ++i) {
      if (i > 0 && cum > 0.88f * Z) { keep = i; break; }
      cum += red[i];
    }
    float best = -1e30f;
    int btok = cand_i[0];
    for (int i = 0; i < keep; ++i) {
      int tok = cand_i[i];
      unsigned hsh = hash_u32(((unsigned)(t * NB + b) << 16) ^ (unsigned)tok ^ 0x9e3779b9U);
      float u = ((hsh >> 8) + 0.5f) * (1.0f / 16777216.0f);
      float g = cand_v[i] - __logf(-__logf(u));
      if (g > best) { best = g; btok = tok; }
    }
    word[b] = btok;
    hist[t * NB + b] = btok;
    tokens_out[b * NT + t] = btok;
    atomicAdd(&counts[btok], 1.0f);  // integer-valued float sums: order-independent
  }
}

}  // namespace

extern "C" void kernel_launch(void* const* d_in, const int* in_sizes, int n_in,
                              void* d_out, int out_size, void* d_ws, size_t ws_size,
                              hipStream_t stream) {
  (void)in_sizes; (void)n_in; (void)out_size;
  const float* thought = (const float*)d_in[0];
  const float* emb     = (const float*)d_in[1];
  const float* w_ih0   = (const float*)d_in[2];
  const float* w_hh0   = (const float*)d_in[3];
  const float* b_ih0   = (const float*)d_in[4];
  const float* b_hh0   = (const float*)d_in[5];
  const float* w_ih1   = (const float*)d_in[6];
  const float* w_hh1   = (const float*)d_in[7];
  const float* b_ih1   = (const float*)d_in[8];
  const float* b_hh1   = (const float*)d_in[9];
  const float* gw1     = (const float*)d_in[10];
  const float* gb1     = (const float*)d_in[11];
  const float* gw2     = (const float*)d_in[12];
  const float* gb2     = (const float*)d_in[13];
  const float* out_w   = (const float*)d_in[14];
  const float* out_b   = (const float*)d_in[15];
  int* tokens_out = (int*)d_out;

  // ---- workspace carve ----
  char* p = (char*)d_ws;
  auto carve = [&](size_t bytes) -> char* {
    char* r = p; p += (bytes + 255) & ~(size_t)255; return r;
  };
  float* h1      = (float*)carve((size_t)NB * NH * 4);
  float* c1      = (float*)carve((size_t)NB * NH * 4);
  float* h2      = (float*)carve((size_t)NB * NH * 4);
  float* c2      = (float*)carve((size_t)NB * NH * 4);
  float* prevctx = (float*)carve((size_t)NB * NH * 4);
  float* gates   = (float*)carve((size_t)NB * NG * 4);      // reused as tanh-GEMM C
  float* logits  = (float*)carve((size_t)NB * NV * 4);
  float* counts  = (float*)carve((size_t)NV * 4);
  int*   hist    = (int*)carve((size_t)NT * NB * 4);
  int*   word    = (int*)carve((size_t)NB * 4);
  bf16*  x0      = (bf16*)carve((size_t)NB * (NE + NH) * 2);
  bf16*  h1b     = (bf16*)carve((size_t)NB * NH * 2);
  bf16*  h2b     = (bf16*)carve((size_t)NB * NH * 2);
  bf16*  outb    = (bf16*)carve((size_t)NB * NH * 2);
  bf16*  tmpb    = (bf16*)carve((size_t)NB * NH * 2);
  // bf16 weight mirrors (used only if workspace is large enough)
  const size_t n_wih0 = (size_t)NG * (NE + NH);
  const size_t n_whh  = (size_t)NG * NH;
  const size_t n_gw1  = (size_t)NH * NH;
  const size_t n_outw = (size_t)NV * NH;
  bf16* w_ih0b = (bf16*)carve(n_wih0 * 2);
  bf16* w_hh0b = (bf16*)carve(n_whh * 2);
  bf16* w_ih1b = (bf16*)carve(n_whh * 2);
  bf16* w_hh1b = (bf16*)carve(n_whh * 2);
  bf16* gw1b   = (bf16*)carve(n_gw1 * 2);
  bf16* out_wb = (bf16*)carve(n_outw * 2);
  const bool pre = (size_t)(p - (char*)d_ws) <= ws_size;  // fixed per harness: deterministic

  if (pre) {  // one-time bf16 conversion: halves per-step HBM weight traffic
    k_cvt<<<(int)((n_wih0 + 255) / 256), 256, 0, stream>>>(w_ih0, w_ih0b, (int)n_wih0);
    k_cvt<<<(int)((n_whh  + 255) / 256), 256, 0, stream>>>(w_hh0, w_hh0b, (int)n_whh);
    k_cvt<<<(int)((n_whh  + 255) / 256), 256, 0, stream>>>(w_ih1, w_ih1b, (int)n_whh);
    k_cvt<<<(int)((n_whh  + 255) / 256), 256, 0, stream>>>(w_hh1, w_hh1b, (int)n_whh);
    k_cvt<<<(int)((n_gw1  + 255) / 256), 256, 0, stream>>>(gw1,   gw1b,   (int)n_gw1);
    k_cvt<<<(int)((n_outw + 255) / 256), 256, 0, stream>>>(out_w, out_wb, (int)n_outw);
  }

  k_init<<<(NB * NH + 255) / 256, 256, 0, stream>>>(thought, h1, c1, h2, c2,
                                                    prevctx, word, counts, hist);

  const int prep_total = NB * (NE + NH) + 2 * NB * NH;  // 229376
  for (int t = 0; t < NT; ++t) {
    k_prep<<<prep_total / 256, 256, 0, stream>>>(emb, word, prevctx, h1, h2,
                                                 x0, h1b, h2b);
    // LSTM cell 0: gates = x0@w_ih0^T + h1b@w_hh0^T + b_ih0 + b_hh0
    if (pre)
      k_gemm<bf16><<<NG / 32, 256, 0, stream>>>(x0, NE + NH, w_ih0b, h1b, NH, w_hh0b,
                                                b_ih0, b_hh0, nullptr, gates, nullptr, NG, 0);
    else
      k_gemm<float><<<NG / 32, 256, 0, stream>>>(x0, NE + NH, w_ih0, h1b, NH, w_hh0,
                                                 b_ih0, b_hh0, nullptr, gates, nullptr, NG, 0);
    k_lstm_elem<<<NB * NH / 256, 256, 0, stream>>>(gates, h1, c1, h1b);
    // LSTM cell 1
    if (pre)
      k_gemm<bf16><<<NG / 32, 256, 0, stream>>>(h1b, NH, w_ih1b, h2b, NH, w_hh1b,
                                                b_ih1, b_hh1, nullptr, gates, nullptr, NG, 0);
    else
      k_gemm<float><<<NG / 32, 256, 0, stream>>>(h1b, NH, w_ih1, h2b, NH, w_hh1,
                                                 b_ih1, b_hh1, nullptr, gates, nullptr, NG, 0);
    k_lstm_elem<<<NB * NH / 256, 256, 0, stream>>>(gates, h2, c2, h2b);
    // tanh(h2 @ gw1^T + gb1)
    if (pre)
      k_gemm<bf16><<<NH / 32, 256, 0, stream>>>(h2b, NH, gw1b, nullptr, 0, nullptr,
                                                gb1, nullptr, nullptr, gates, tmpb, NH, 1);
    else
      k_gemm<float><<<NH / 32, 256, 0, stream>>>(h2b, NH, gw1, nullptr, 0, nullptr,
                                                 gb1, nullptr, nullptr, gates, tmpb, NH, 1);
    // gate scalar + out = h2*gate (feeds prev_ctx and logits GEMM)
    k_gatedot<<<NB, 256, 0, stream>>>(tmpb, gw2, gb2, h2, prevctx, outb);
    // logits GEMM with fused count-penalty / temperature / count-bonus epilogue
    if (pre)
      k_gemm<bf16><<<NV / 32, 256, 0, stream>>>(outb, NH, out_wb, nullptr, 0, nullptr,
                                                out_b, nullptr, counts, logits, nullptr, NV, 2);
    else
      k_gemm<float><<<NV / 32, 256, 0, stream>>>(outb, NH, out_w, nullptr, 0, nullptr,
                                                 out_b, nullptr, counts, logits, nullptr, NV, 2);
    k_hist_pen<<<1, NB, 0, stream>>>(logits, hist, t);
    k_sample<<<NB, 256, 0, stream>>>(logits, counts, hist, word, tokens_out, t);
  }
}
